// SwinTransformerBlockPost_77644418777282
// MI455X (gfx1250) — matrix operand order repaired and checked by reading, not compiled, not run on hardware
//
#include <hip/hip_runtime.h>

// ---------------------------------------------------------------------------
// Types / WMMA helpers (gfx1250 wave32 WMMA, f16 inputs, f32 accum)
// ---------------------------------------------------------------------------
typedef _Float16 half_t;
typedef __attribute__((ext_vector_type(16))) _Float16 v16h;
typedef __attribute__((ext_vector_type(8)))  float    v8f;
typedef __attribute__((ext_vector_type(4)))  unsigned int u32x4;

union Frag16 { v16h h; u32x4 u[2]; };

// A-matrix fragment (16x32 f16): lane%16 = row M, K = k0 + 8*(lane/16) + [0..7]
// and K = k0 + 16 + 8*(lane/16) + [0..7]  (ISA 7.12.2, 16-bit A 16x32)
__device__ inline v16h load_a(const half_t* base, int ldk, int m0, int k0, int lane) {
  int m  = m0 + (lane & 15);
  int kk = k0 + ((lane >> 4) << 3);
  Frag16 f;
  f.u[0] = *(const u32x4*)(base + m * ldk + kk);
  f.u[1] = *(const u32x4*)(base + m * ldk + kk + 16);
  return f.h;
}

// B-matrix fragment (32x16 f16) loaded from B^T stored [n][k]:
// lane%16 = column n, K = k0 + 8*(lane/16) + [0..7] and K = k0 + 16 + 8*(lane/16) + [0..7]
__device__ inline v16h load_b(const half_t* baseT, int ldk, int n0, int k0, int lane) {
  int n  = n0 + (lane & 15);
  int kk = k0 + ((lane >> 4) << 3);
  const u32x4* p = (const u32x4*)(baseT + n * ldk + kk);
  Frag16 f;
  f.u[0] = p[0];
  f.u[1] = p[2];
  return f.h;
}

__device__ inline v8f wmma16(v16h a, v16h b, v8f c) {
  // (neg_a, A, neg_b, B, c_mod, C, reuse_a, reuse_b)
  return __builtin_amdgcn_wmma_f32_16x16x32_f16(false, a, false, b, (short)0, c, false, false);
}

// ---------------------------------------------------------------------------
// Constants
// ---------------------------------------------------------------------------
#define WS7    7
#define NTOK   49            // tokens per window
#define CDIM   128
#define HEADS  4
#define HD     32
#define HID    512
#define IMGB   64
#define IMGHW  56
#define NWIN   64            // windows per image (8x8)

// workspace byte offsets (all < 512 KB)
#define WSO_BIAS   0                         // f32 [4*49*49]          38,416 B
#define WSO_SCALE  40960                     // f32 [4]
#define WSO_QKVT   65536                     // f16 [384*128]          98,304 B
#define WSO_PROJT  (65536 + 98304)           // f16 [128*128]          32,768 B
#define WSO_FC1T   (65536 + 98304 + 32768)   // f16 [512*128]         131,072 B
#define WSO_FC2T   (65536 + 98304 + 32768 + 131072) // f16 [128*512]  131,072 B
#define WS_BYTES   (65536 + 98304 + 32768 + 131072 + 131072)

// attention kernel dynamic-LDS layout (bytes)
#define AO_XA   0                 // f16 64x144 : x window, later qn
#define AO_Q    18432             // f32 64x136 : q raw; later S f32 64x72; later proj-out
#define AO_K    (18432+34816)     // f32 64x136 : k raw; later P f16 64x80 + of f16 64x144
#define AO_KN   (AO_K+34816)      // f16 64x144 : kn; later LN stats
#define AO_VT   (AO_KN+18432)     // f16 4x32x72 : v transposed per head
#define ATTN_SMEM (AO_VT+18432)   // 124,928 B

// mlp kernel dynamic-LDS layout (bytes)
#define MO_YH   0                 // f16 64x144
#define MO_Y32  18432             // f32 64x136
#define MO_H    (18432+34816)     // f16 64x528
#define MO_G    (MO_H+67584)      // f32 64x136
#define MO_ST   (MO_G+34816)      // f32 stats 2x64
#define MLP_SMEM (MO_ST+512)      // 156,160 B

// ---------------------------------------------------------------------------
// Kernel 1: convert + transpose GEMM weights to f16  W^T[n][k]
// ---------------------------------------------------------------------------
__global__ __launch_bounds__(256) void prep_weights(
    const float* __restrict__ qkv_w, const float* __restrict__ proj_w,
    const float* __restrict__ fc1_w, const float* __restrict__ fc2_w,
    half_t* __restrict__ qkvT, half_t* __restrict__ projT,
    half_t* __restrict__ fc1T, half_t* __restrict__ fc2T)
{
  int i = blockIdx.x * 256 + threadIdx.x;
  if (i < 49152) {                       // qkv_w [128,384] -> [384][128]
    int n = i >> 7, k = i & 127;
    qkvT[i] = (half_t)qkv_w[k * 384 + n];
  } else if (i < 65536) {                // proj_w [128,128] -> [128][128]
    int j = i - 49152;
    int n = j >> 7, k = j & 127;
    projT[j] = (half_t)proj_w[k * 128 + n];
  } else if (i < 131072) {               // fc1_w [128,512] -> [512][128]
    int j = i - 65536;
    int n = j >> 7, k = j & 127;
    fc1T[j] = (half_t)fc1_w[k * 512 + n];
  } else if (i < 196608) {               // fc2_w [512,128] -> [128][512]
    int j = i - 131072;
    int n = j >> 9, k = j & 511;
    fc2T[j] = (half_t)fc2_w[k * 128 + n];
  }
}

// ---------------------------------------------------------------------------
// Kernel 2: continuous relative-position bias table + per-head scale
// ---------------------------------------------------------------------------
__global__ __launch_bounds__(256) void prep_bias(
    const float* __restrict__ rpe_w1, const float* __restrict__ rpe_b1,
    const float* __restrict__ rpe_w2, const float* __restrict__ logit_scale,
    float* __restrict__ bias16, float* __restrict__ scale)
{
  __shared__ float tbl[169][4];
  int tid = threadIdx.x;
  if (tid < 169) {
    int i = tid / 13, j = tid % 13;      // entry = i*13 + j; t = (h[j], h[i])
    float d0 = (float)(j - 6) * (8.0f / 36.0f);
    float d1 = (float)(i - 6) * (8.0f / 36.0f);
    const float invlog8 = 0.48089834696298783f;  // 1/ln(8)
    float t0 = copysignf(log1pf(fabsf(d0)), d0) * invlog8;
    float t1 = copysignf(log1pf(fabsf(d1)), d1) * invlog8;
    float a0 = 0.f, a1 = 0.f, a2 = 0.f, a3 = 0.f;
    for (int h = 0; h < 512; ++h) {
      float hid = fmaxf(t0 * rpe_w1[h] + t1 * rpe_w1[512 + h] + rpe_b1[h], 0.f);
      a0 += hid * rpe_w2[h * 4 + 0];
      a1 += hid * rpe_w2[h * 4 + 1];
      a2 += hid * rpe_w2[h * 4 + 2];
      a3 += hid * rpe_w2[h * 4 + 3];
    }
    tbl[tid][0] = a0; tbl[tid][1] = a1; tbl[tid][2] = a2; tbl[tid][3] = a3;
  }
  if (tid < 4) scale[tid] = __expf(fminf(logit_scale[tid], logf(100.0f)));
  __syncthreads();
  for (int e = tid; e < HEADS * NTOK * NTOK; e += 256) {
    int hh = e / (NTOK * NTOK), rem = e % (NTOK * NTOK);
    int qi = rem / NTOK, kj = rem % NTOK;
    int dc = (qi % 7) - (kj % 7);        // rel[...,0] = dcol
    int dr = (qi / 7) - (kj / 7);        // rel[...,1] = drow
    int idx = (dc + 6) * 13 + (dr + 6);
    float v = tbl[idx][hh];
    bias16[e] = 16.0f / (1.0f + __expf(-v));
  }
}

// ---------------------------------------------------------------------------
// Kernel 3: fused shifted-window attention + proj + residual + LayerNorm1
//           one block (8 waves) per window; writes y into y_out (= d_out)
// ---------------------------------------------------------------------------
__global__ __launch_bounds__(256) void attn_kernel(
    const float* __restrict__ x, const float* __restrict__ mask,
    const float* __restrict__ n1g, const float* __restrict__ n1b,
    const float* __restrict__ qb, const float* __restrict__ vb,
    const half_t* __restrict__ qkvT, const half_t* __restrict__ projT,
    const float* __restrict__ projb, const float* __restrict__ bias16,
    const float* __restrict__ scale, float* __restrict__ y_out)
{
  extern __shared__ char smem[];
  half_t* s_xa = (half_t*)(smem + AO_XA);          // xa, then qn
  float*  s_q  = (float*)(smem + AO_Q);
  float*  s_k  = (float*)(smem + AO_K);
  half_t* s_kn = (half_t*)(smem + AO_KN);
  half_t* s_vT = (half_t*)(smem + AO_VT);          // [head][32][72]
  float*  s_S  = (float*)(smem + AO_Q);            // 64x72 (q dead)
  half_t* s_P  = (half_t*)(smem + AO_K);           // 64x80 (k dead)
  half_t* s_of = (half_t*)(smem + AO_K + 16384);   // 64x144
  float*  s_po = (float*)(smem + AO_Q);            // 64x136 (S dead)
  float*  s_mu = (float*)(smem + AO_KN);           // 64 (kn dead)
  float*  s_rs = (float*)(smem + AO_KN + 256);     // 64

  const int tid = threadIdx.x, lane = tid & 31, wv = tid >> 5;
  const int w = blockIdx.x;
  const int b = w >> 6, wloc = w & 63, wy = wloc >> 3, wx = wloc & 7;

  // ---- phase 1: gather shifted window -> LDS f16 (rows 49..63 zero) ----
  for (int e = tid; e < 64 * CDIM; e += 256) {
    int m = e >> 7, c = e & 127;
    half_t val = (half_t)0.f;
    if (m < NTOK) {
      int r = m / 7, cc = m % 7;
      int hr = wy * 7 + r + 3;  if (hr >= IMGHW) hr -= IMGHW;
      int wc = wx * 7 + cc + 3; if (wc >= IMGHW) wc -= IMGHW;
      val = (half_t)x[(((b * IMGHW + hr) * IMGHW) + wc) * CDIM + c];
    }
    s_xa[m * 144 + c] = val;
  }
  __syncthreads();

  // ---- phase 2: QKV = xa @ qkvT^T + bias ; 4x24 tiles, 12/wave ----
  for (int t = 0; t < 12; ++t) {
    int tile = wv * 12 + t;
    int mt = tile / 24, nt = tile % 24;
    int m0 = mt * 16, n0 = nt * 16;
    v8f acc = {};
    for (int k0 = 0; k0 < CDIM; k0 += 32)
      acc = wmma16(load_a(s_xa, 144, m0, k0, lane), load_b(qkvT, 128, n0, k0, lane), acc);
    int n = n0 + (lane & 15);
    int mbase = m0 + ((lane >> 4) << 3);
    if (n < 128) {                 // q (+q_bias)
      float bias = qb[n];
      for (int r = 0; r < 8; ++r) s_q[(mbase + r) * 136 + n] = acc[r] + bias;
    } else if (n < 256) {          // k (no bias)
      int nk = n - 128;
      for (int r = 0; r < 8; ++r) s_k[(mbase + r) * 136 + nk] = acc[r];
    } else {                       // v (+v_bias) stored transposed per head
      int nv = n - 256, hh = nv >> 5, cc = nv & 31;
      float bias = vb[nv];
      half_t* vt = s_vT + (hh * 32 + cc) * 72;
      for (int r = 0; r < 8; ++r) vt[mbase + r] = (half_t)(acc[r] + bias);
    }
  }
  __syncthreads();

  // ---- phase 3: cosine-normalize q,k per (token, head) -> f16 ----
  {
    int m = tid >> 2, hh = tid & 3;
    const float* qrow = s_q + m * 136 + hh * HD;
    float ss = 0.f;
    for (int c = 0; c < HD; ++c) { float v = qrow[c]; ss += v * v; }
    float rn = rsqrtf(fmaxf(ss, 1e-12f));
    half_t* qn = s_xa + m * 144 + hh * HD;         // reuse xa region
    for (int c = 0; c < HD; ++c) qn[c] = (half_t)(qrow[c] * rn);

    const float* krow = s_k + m * 136 + hh * HD;
    ss = 0.f;
    for (int c = 0; c < HD; ++c) { float v = krow[c]; ss += v * v; }
    rn = rsqrtf(fmaxf(ss, 1e-12f));
    half_t* kn = s_kn + m * 144 + hh * HD;
    for (int c = 0; c < HD; ++c) kn[c] = (half_t)(krow[c] * rn);
  }
  __syncthreads();

  // ---- phase 4: per-head attention ----
  for (int hh = 0; hh < HEADS; ++hh) {
    // S = qn_h @ kn_h^T  (4x4 tiles, K=32 -> one WMMA each; 2 tiles/wave)
    for (int t = 0; t < 2; ++t) {
      int tile = wv * 2 + t;
      int mt = tile >> 2, nt = tile & 3;
      int m0 = mt * 16, n0 = nt * 16;
      v8f acc = {};
      acc = wmma16(load_a(s_xa, 144, m0, hh * HD, lane),
                   load_b(s_kn, 144, n0, hh * HD, lane), acc);
      int n = n0 + (lane & 15);
      int mbase = m0 + ((lane >> 4) << 3);
      float sc = scale[hh];
      for (int r = 0; r < 8; ++r) {
        int m = mbase + r;
        float v;
        if (n < NTOK) {
          v = acc[r] * sc;
          if (m < NTOK)
            v += bias16[(hh * NTOK + m) * NTOK + n] + mask[(wloc * NTOK + m) * NTOK + n];
        } else {
          v = -1e30f;
        }
        s_S[m * 72 + n] = v;
      }
    }
    __syncthreads();

    // softmax over 49 keys: 4 threads per row (quad shuffle reduction)
    {
      int m = tid >> 2, qq = tid & 3;
      float mx = -1e30f;
      for (int n = qq; n < NTOK; n += 4) mx = fmaxf(mx, s_S[m * 72 + n]);
      mx = fmaxf(mx, __shfl_xor(mx, 1, 32));
      mx = fmaxf(mx, __shfl_xor(mx, 2, 32));
      float ex[13];
      int cnt = 0;
      float sum = 0.f;
      for (int n = qq; n < NTOK; n += 4) {
        float e0 = __expf(s_S[m * 72 + n] - mx);
        ex[cnt++] = e0;
        sum += e0;
      }
      sum += __shfl_xor(sum, 1, 32);
      sum += __shfl_xor(sum, 2, 32);
      float inv = 1.f / sum;
      cnt = 0;
      for (int n = qq; n < 64; n += 4) {
        half_t pv = (half_t)0.f;
        if (n < NTOK) pv = (half_t)(ex[cnt++] * inv);
        s_P[m * 80 + n] = pv;       // K padded to 64 with zeros
      }
    }
    __syncthreads();

    // O_h = P @ V_h  (4x2 tiles, K=64 -> 2 WMMA; 1 tile/wave)
    {
      int mt = wv >> 1, nt = wv & 1;
      int m0 = mt * 16, n0 = nt * 16;
      v8f acc = {};
      const half_t* vtb = s_vT + hh * HD * 72;
      for (int k0 = 0; k0 < 64; k0 += 32)
        acc = wmma16(load_a(s_P, 80, m0, k0, lane), load_b(vtb, 72, n0, k0, lane), acc);
      int c = n0 + (lane & 15);
      int mbase = m0 + ((lane >> 4) << 3);
      for (int r = 0; r < 8; ++r)
        s_of[(mbase + r) * 144 + hh * HD + c] = (half_t)acc[r];
    }
    __syncthreads();
  }

  // ---- phase 5: proj  (4x8 tiles, 4/wave) ----
  for (int t = 0; t < 4; ++t) {
    int tile = wv * 4 + t;
    int mt = tile >> 3, nt = tile & 7;
    int m0 = mt * 16, n0 = nt * 16;
    v8f acc = {};
    for (int k0 = 0; k0 < CDIM; k0 += 32)
      acc = wmma16(load_a(s_of, 144, m0, k0, lane), load_b(projT, 128, n0, k0, lane), acc);
    int n = n0 + (lane & 15);
    int mbase = m0 + ((lane >> 4) << 3);
    float bias = projb[n];
    for (int r = 0; r < 8; ++r) s_po[(mbase + r) * 136 + n] = acc[r] + bias;
  }
  __syncthreads();

  // ---- phase 6: y = x + LN(proj_out), scattered back through un-shift ----
  if (tid < 64) {
    int m = tid;
    float sum = 0.f;
    for (int c = 0; c < CDIM; ++c) sum += s_po[m * 136 + c];
    float mu = sum * (1.f / 128.f);
    float vs = 0.f;
    for (int c = 0; c < CDIM; ++c) { float d = s_po[m * 136 + c] - mu; vs += d * d; }
    s_mu[m] = mu;
    s_rs[m] = rsqrtf(vs * (1.f / 128.f) + 1e-3f);
  }
  __syncthreads();
  for (int e = tid; e < NTOK * CDIM; e += 256) {
    int m = e >> 7, c = e & 127;
    int r = m / 7, cc = m % 7;
    int hr = wy * 7 + r + 3;  if (hr >= IMGHW) hr -= IMGHW;
    int wc = wx * 7 + cc + 3; if (wc >= IMGHW) wc -= IMGHW;
    long idx = (((long)(b * IMGHW + hr)) * IMGHW + wc) * CDIM + c;
    y_out[idx] = x[idx] + (s_po[m * 136 + c] - s_mu[m]) * s_rs[m] * n1g[c] + n1b[c];
  }
}

// ---------------------------------------------------------------------------
// Kernel 4: fused MLP (FC1 + exact GELU + FC2) + residual + LayerNorm2
//           64 tokens per block; reads & writes d_out in place (per-block safe)
// ---------------------------------------------------------------------------
__global__ __launch_bounds__(256) void mlp_kernel(
    float* __restrict__ y, const half_t* __restrict__ fc1T,
    const float* __restrict__ fc1b, const half_t* __restrict__ fc2T,
    const float* __restrict__ fc2b, const float* __restrict__ n2g,
    const float* __restrict__ n2b)
{
  extern __shared__ char smem[];
  half_t* s_yh  = (half_t*)(smem + MO_YH);   // 64x144 f16
  float*  s_y32 = (float*)(smem + MO_Y32);   // 64x136 f32
  half_t* s_h   = (half_t*)(smem + MO_H);    // 64x528 f16 (hidden, K pad n/a)
  float*  s_g   = (float*)(smem + MO_G);     // 64x136 f32
  float*  s_mu  = (float*)(smem + MO_ST);
  float*  s_rs  = (float*)(smem + MO_ST + 256);

  const int tid = threadIdx.x, lane = tid & 31, wv = tid >> 5;
  const long base = (long)blockIdx.x * 64 * CDIM;

  for (int e = tid; e < 64 * CDIM; e += 256) {
    float v = y[base + e];
    int m = e >> 7, c = e & 127;
    s_y32[m * 136 + c] = v;
    s_yh[m * 144 + c] = (half_t)v;
  }
  __syncthreads();

  // FC1: [64x128] @ [128x512] ; 4x32 tiles, 16/wave ; + bias + exact GELU
  for (int t = 0; t < 16; ++t) {
    int tile = wv * 16 + t;
    int mt = tile >> 5, nt = tile & 31;
    int m0 = mt * 16, n0 = nt * 16;
    v8f acc = {};
    for (int k0 = 0; k0 < CDIM; k0 += 32)
      acc = wmma16(load_a(s_yh, 144, m0, k0, lane), load_b(fc1T, 128, n0, k0, lane), acc);
    int n = n0 + (lane & 15);
    int mbase = m0 + ((lane >> 4) << 3);
    float bias = fc1b[n];
    for (int r = 0; r < 8; ++r) {
      float v = acc[r] + bias;
      v = 0.5f * v * (1.f + erff(v * 0.70710678118654752f));
      s_h[(mbase + r) * 528 + n] = (half_t)v;
    }
  }
  __syncthreads();

  // FC2: [64x512] @ [512x128] ; 4x8 tiles, 4/wave ; K=512 -> 16 WMMA steps
  for (int t = 0; t < 4; ++t) {
    int tile = wv * 4 + t;
    int mt = tile >> 3, nt = tile & 7;
    int m0 = mt * 16, n0 = nt * 16;
    v8f acc = {};
    for (int k0 = 0; k0 < HID; k0 += 32)
      acc = wmma16(load_a(s_h, 528, m0, k0, lane), load_b(fc2T, 512, n0, k0, lane), acc);
    int n = n0 + (lane & 15);
    int mbase = m0 + ((lane >> 4) << 3);
    float bias = fc2b[n];
    for (int r = 0; r < 8; ++r) s_g[(mbase + r) * 136 + n] = acc[r] + bias;
  }
  __syncthreads();

  if (tid < 64) {
    int m = tid;
    float sum = 0.f;
    for (int c = 0; c < CDIM; ++c) sum += s_g[m * 136 + c];
    float mu = sum * (1.f / 128.f);
    float vs = 0.f;
    for (int c = 0; c < CDIM; ++c) { float d = s_g[m * 136 + c] - mu; vs += d * d; }
    s_mu[m] = mu;
    s_rs[m] = rsqrtf(vs * (1.f / 128.f) + 1e-3f);
  }
  __syncthreads();
  for (int e = tid; e < 64 * CDIM; e += 256) {
    int m = e >> 7, c = e & 127;
    y[base + e] = s_y32[m * 136 + c] + (s_g[m * 136 + c] - s_mu[m]) * s_rs[m] * n2g[c] + n2b[c];
  }
}

// ---------------------------------------------------------------------------
// launch
// ---------------------------------------------------------------------------
extern "C" void kernel_launch(void* const* d_in, const int* in_sizes, int n_in,
                              void* d_out, int out_size, void* d_ws, size_t ws_size,
                              hipStream_t stream) {
  (void)in_sizes; (void)n_in; (void)out_size; (void)ws_size;

  const float* x           = (const float*)d_in[0];
  const float* mask        = (const float*)d_in[1];
  const float* n1g         = (const float*)d_in[2];
  const float* n1b         = (const float*)d_in[3];
  const float* qkv_w       = (const float*)d_in[4];
  const float* q_bias      = (const float*)d_in[5];
  const float* v_bias      = (const float*)d_in[6];
  const float* logit_scale = (const float*)d_in[7];
  const float* rpe_w1      = (const float*)d_in[8];
  const float* rpe_b1      = (const float*)d_in[9];
  const float* rpe_w2      = (const float*)d_in[10];
  const float* proj_w      = (const float*)d_in[11];
  const float* proj_b      = (const float*)d_in[12];
  const float* n2g         = (const float*)d_in[13];
  const float* n2b         = (const float*)d_in[14];
  const float* fc1_w       = (const float*)d_in[15];
  const float* fc1_b       = (const float*)d_in[16];
  const float* fc2_w       = (const float*)d_in[17];
  const float* fc2_b       = (const float*)d_in[18];

  char* ws = (char*)d_ws;
  float*  ws_bias  = (float*)(ws + WSO_BIAS);
  float*  ws_scale = (float*)(ws + WSO_SCALE);
  half_t* ws_qkvT  = (half_t*)(ws + WSO_QKVT);
  half_t* ws_projT = (half_t*)(ws + WSO_PROJT);
  half_t* ws_fc1T  = (half_t*)(ws + WSO_FC1T);
  half_t* ws_fc2T  = (half_t*)(ws + WSO_FC2T);
  float*  out      = (float*)d_out;   // doubles as intermediate y

  // allow >64KB dynamic LDS (metadata-only host call; deterministic, capture-safe)
  (void)hipFuncSetAttribute(reinterpret_cast<const void*>(attn_kernel),
                            hipFuncAttributeMaxDynamicSharedMemorySize, ATTN_SMEM);
  (void)hipFuncSetAttribute(reinterpret_cast<const void*>(mlp_kernel),
                            hipFuncAttributeMaxDynamicSharedMemorySize, MLP_SMEM);

  prep_weights<<<768, 256, 0, stream>>>(qkv_w, proj_w, fc1_w, fc2_w,
                                        ws_qkvT, ws_projT, ws_fc1T, ws_fc2T);
  prep_bias<<<1, 256, 0, stream>>>(rpe_w1, rpe_b1, rpe_w2, logit_scale,
                                   ws_bias, ws_scale);
  attn_kernel<<<IMGB * NWIN, 256, ATTN_SMEM, stream>>>(
      x, mask, n1g, n1b, q_bias, v_bias, ws_qkvT, ws_projT, proj_b,
      ws_bias, ws_scale, out);
  mlp_kernel<<<(IMGB * IMGHW * IMGHW) / 64, 256, MLP_SMEM, stream>>>(
      out, ws_fc1T, fc1_b, ws_fc2T, fc2_b, n2g, n2b);
}
